// SelfAttentionHead_2259152797990
// MI455X (gfx1250) — compile-verified, matrix operand order checked by reading, not run on hardware
//
#include <hip/hip_runtime.h>
#include <hip/hip_bf16.h>

// ---------------------------------------------------------------------------
// SelfAttentionHead for MI455X (gfx1250): bf16 WMMA flash-attention pipeline.
//   B=8, T=2048, C=1024, DK=64
//   out = softmax((x Wq)(x Wk)^T * dk^-0.5) (x Wv)
// ---------------------------------------------------------------------------

typedef __attribute__((ext_vector_type(16))) __bf16 v16bf;
typedef __attribute__((ext_vector_type(2)))  __bf16 v2bf;
typedef __attribute__((ext_vector_type(8)))  float  v8f;
typedef __attribute__((ext_vector_type(4)))  int    v4i;

#define AS1 __attribute__((address_space(1)))
#define AS3 __attribute__((address_space(3)))

union ABu {
    v16bf v;
    unsigned int ui[8];
    unsigned short us[16];
};

#define ATT_B  8
#define ATT_T  2048
#define ATT_C  1024
#define ATT_DK 64

#if __has_builtin(__builtin_amdgcn_global_load_async_to_lds_b128) && \
    __has_builtin(__builtin_amdgcn_s_wait_asynccnt)
#define HAS_ASYNC_LDS 1
#else
#define HAS_ASYNC_LDS 0
#endif

// ---------------- bf16 conversion helpers ----------------------------------
#if __has_builtin(__builtin_amdgcn_cvt_pk_bf16_f32)
__device__ __forceinline__ unsigned int pack_bf16x2(float a, float b) {
    v2bf r = __builtin_amdgcn_cvt_pk_bf16_f32(a, b);
    return __builtin_bit_cast(unsigned int, r);
}
__device__ __forceinline__ unsigned short f32_to_bf16(float f) {
    v2bf r = __builtin_amdgcn_cvt_pk_bf16_f32(f, 0.0f);
    return (unsigned short)(__builtin_bit_cast(unsigned int, r) & 0xFFFFu);
}
#else
// Native fptrunc f32->bf16: lets the backend select hardware cvt on gfx1250.
__device__ __forceinline__ unsigned int pack_bf16x2(float a, float b) {
    v2bf r;
    r[0] = (__bf16)a;
    r[1] = (__bf16)b;
    return __builtin_bit_cast(unsigned int, r);
}
__device__ __forceinline__ unsigned short f32_to_bf16(float f) {
    __bf16 h = (__bf16)f;
    return __builtin_bit_cast(unsigned short, h);
}
#endif

// exp2 (v_exp_f32 is a native exp2)
#if __has_builtin(__builtin_amdgcn_exp2f)
#define EXP2F(x) __builtin_amdgcn_exp2f(x)
#else
#define EXP2F(x) exp2f(x)
#endif

__device__ __forceinline__ v8f wmma_bf16(v16bf a, v16bf b, v8f c) {
    // v_wmma_f32_16x16x32_bf16  (neg_a, A, neg_b, B, c_mod, C, reuse_a, reuse_b)
    return __builtin_amdgcn_wmma_f32_16x16x32_bf16(false, a, false, b, (short)0, c, false, false);
}

// Load a 16x32 bf16 tile (row-major, `stride` elems between rows) into the
// wave32 A/B operand layout:
//   lane L: row = L&15, kBase = (L>>4)*8; holds K = kBase..kBase+7 and
//   kBase+16..kBase+23 as packed bf16 pairs  -> two b128 loads per lane.
__device__ __forceinline__ v16bf load_tile(const unsigned short* __restrict__ tile0,
                                           int stride) {
    const int lane = threadIdx.x & 31;
    const int r    = lane & 15;
    const int kb   = (lane >> 4) << 3;          // 0 or 8
    const unsigned short* p = tile0 + (size_t)r * stride + kb;
    ABu t;
    *(uint4*)&t.ui[0] = *(const uint4*)(p);
    *(uint4*)&t.ui[4] = *(const uint4*)(p + 16);
    return t.v;
}

// ---------------------------------------------------------------------------
// Kernel 0: transpose weights to Wt[192][1024] bf16.
// ---------------------------------------------------------------------------
__global__ void k_transpose_w(const float* __restrict__ Wq,
                              const float* __restrict__ Wk,
                              const float* __restrict__ Wv,
                              unsigned short* __restrict__ Wt) {
    const int n   = blockIdx.x;                 // 0..191
    const int mat = n >> 6;
    const int nn  = n & 63;
    const float* src = (mat == 0) ? Wq : (mat == 1) ? Wk : Wv;
    for (int k = threadIdx.x; k < ATT_C; k += blockDim.x)
        Wt[(size_t)n * ATT_C + k] = f32_to_bf16(src[(size_t)k * ATT_DK + nn]);
}

// ---------------------------------------------------------------------------
// Kernel 1: QKV projection -> Q,K row-major bf16 and V^T [b][d][t] bf16.
// ---------------------------------------------------------------------------
__global__ void k_qkv(const float* __restrict__ x,
                      const unsigned short* __restrict__ Wt,
                      unsigned short* __restrict__ Qb,
                      unsigned short* __restrict__ Kb,
                      unsigned short* __restrict__ VT) {
    __shared__ __align__(16) unsigned int ldsx[16 * 20];

    const int wave = threadIdx.x >> 5;
    const int lane = threadIdx.x & 31;
    const int tok0 = blockIdx.x * 16;
    const int tt0  = wave * 3;                  // first N-tile of this wave

    v8f acc0 = {}, acc1 = {}, acc2 = {};

    for (int kc = 0; kc < ATT_C / 32; ++kc) {
        __syncthreads();
        {   // stage x[16 rows][32 cols] f32 -> bf16 LDS, 4 elems/thread
            const int row  = threadIdx.x >> 3;
            const int col0 = (threadIdx.x & 7) * 4;
            const float4 f = *(const float4*)(x + (size_t)(tok0 + row) * ATT_C
                                                + kc * 32 + col0);
            ldsx[row * 20 + (col0 >> 1)]     = pack_bf16x2(f.x, f.y);
            ldsx[row * 20 + (col0 >> 1) + 1] = pack_bf16x2(f.z, f.w);
        }
        __syncthreads();

        ABu a;
        {   // A tile from LDS in WMMA layout
            const int r   = lane & 15;
            const int kb2 = (lane >> 4) * 4;    // uint offset: kBase/2
            *(uint4*)&a.ui[0] = *(const uint4*)&ldsx[r * 20 + kb2];
            *(uint4*)&a.ui[4] = *(const uint4*)&ldsx[r * 20 + kb2 + 8];
        }
        const v16bf b0 = load_tile(Wt + (size_t)(tt0 + 0) * 16 * ATT_C + kc * 32, ATT_C);
        const v16bf b1 = load_tile(Wt + (size_t)(tt0 + 1) * 16 * ATT_C + kc * 32, ATT_C);
        const v16bf b2 = load_tile(Wt + (size_t)(tt0 + 2) * 16 * ATT_C + kc * 32, ATT_C);
        acc0 = wmma_bf16(a.v, b0, acc0);
        acc1 = wmma_bf16(a.v, b1, acc1);
        acc2 = wmma_bf16(a.v, b2, acc2);
    }

    const int n     = lane & 15;
    const int mBase = (lane >> 4) * 8;
#pragma unroll
    for (int j = 0; j < 3; ++j) {
        const v8f c  = (j == 0) ? acc0 : (j == 1) ? acc1 : acc2;
        const int tt = tt0 + j;
        const int mat = tt >> 2;                // 0=Q, 1=K, 2=V
        const int d   = (tt & 3) * 16 + n;
        if (mat == 2) {
            const int bb = tok0 >> 11;
            const int tl = (tok0 & 2047) + mBase;
            uint4 pk;
            pk.x = pack_bf16x2(c[0], c[1]);
            pk.y = pack_bf16x2(c[2], c[3]);
            pk.z = pack_bf16x2(c[4], c[5]);
            pk.w = pack_bf16x2(c[6], c[7]);
            *(uint4*)(VT + (size_t)(bb * ATT_DK + d) * ATT_T + tl) = pk;
        } else {
            unsigned short* dst = (mat == 0) ? Qb : Kb;
#pragma unroll
            for (int i = 0; i < 8; ++i)
                dst[(size_t)(tok0 + mBase + i) * ATT_DK + d] = f32_to_bf16(c[i]);
        }
    }
}

// ---------------------------------------------------------------------------
// Kernel 2: flash attention.  One block = 4 waves = 64 queries of one batch.
// K / V^T chunks shared via LDS, double-buffered async global->LDS copies.
// ---------------------------------------------------------------------------
#define SK_STRIDE 72   // bf16 elems; 144B row stride -> conflict-free b128
#define SV_STRIDE 40   // bf16 elems;  80B row stride -> conflict-free b128

__device__ __forceinline__ void stage_chunk(const unsigned short* __restrict__ kbb,
                                            const unsigned short* __restrict__ vtb,
                                            int kt,
                                            unsigned short* kbuf,
                                            unsigned short* vbuf,
                                            int tid) {
#if HAS_ASYNC_LDS
#pragma unroll
    for (int i = 0; i < 2; ++i) {               // K: 32 rows x 8 x 16B segs
        const int seg = tid + i * 128;
        const int row = seg >> 3, c16 = seg & 7;
        const unsigned short* g = kbb + (size_t)(kt * 32 + row) * ATT_DK + c16 * 8;
        __builtin_amdgcn_global_load_async_to_lds_b128(
            (AS1 v4i*)g, (AS3 v4i*)(kbuf + row * SK_STRIDE + c16 * 8), 0, 0);
    }
#pragma unroll
    for (int i = 0; i < 2; ++i) {               // V^T: 64 rows x 4 x 16B segs
        const int seg = tid + i * 128;
        const int row = seg >> 2, c16 = seg & 3;
        const unsigned short* g = vtb + (size_t)row * ATT_T + kt * 32 + c16 * 8;
        __builtin_amdgcn_global_load_async_to_lds_b128(
            (AS1 v4i*)g, (AS3 v4i*)(vbuf + row * SV_STRIDE + c16 * 8), 0, 0);
    }
#else
#pragma unroll
    for (int i = 0; i < 2; ++i) {
        const int seg = tid + i * 128;
        const int row = seg >> 3, c16 = seg & 7;
        const uint4 d = *(const uint4*)(kbb + (size_t)(kt * 32 + row) * ATT_DK + c16 * 8);
        *(uint4*)(kbuf + row * SK_STRIDE + c16 * 8) = d;
    }
#pragma unroll
    for (int i = 0; i < 2; ++i) {
        const int seg = tid + i * 128;
        const int row = seg >> 2, c16 = seg & 3;
        const uint4 d = *(const uint4*)(vtb + (size_t)row * ATT_T + kt * 32 + c16 * 8);
        *(uint4*)(vbuf + row * SV_STRIDE + c16 * 8) = d;
    }
#endif
}

__device__ __forceinline__ void wait_stage() {
#if HAS_ASYNC_LDS
    __builtin_amdgcn_s_wait_asynccnt(0);
#endif
}

__global__ void k_attn(const unsigned short* __restrict__ Qb,
                       const unsigned short* __restrict__ Kb,
                       const unsigned short* __restrict__ VT,
                       float* __restrict__ out) {
    __shared__ __align__(16) unsigned short sK[2][32 * SK_STRIDE];
    __shared__ __align__(16) unsigned short sV[2][64 * SV_STRIDE];

    const int tid   = threadIdx.x;
    const int wave  = tid >> 5;
    const int lane  = tid & 31;
    const int qt    = blockIdx.x * 4 + wave;    // all 4 waves in same batch
    const int bb    = qt >> 7;
    const int q0    = (qt & 127) * 16;
    const int qtok0 = bb * ATT_T + q0;

    const unsigned short* kbb = Kb + (size_t)bb * ATT_T * ATT_DK;
    const unsigned short* vtb = VT + (size_t)bb * ATT_DK * ATT_T;

    const v16bf bq0 = load_tile(Qb + (size_t)qtok0 * ATT_DK + 0,  ATT_DK);
    const v16bf bq1 = load_tile(Qb + (size_t)qtok0 * ATT_DK + 32, ATT_DK);

    v8f o0 = {}, o1 = {}, o2 = {}, o3 = {};
    // softmax state kept in log2 domain: exp(x) -> exp2(x * log2e)
    const float cl = 0.125f * 1.44269504088896340736f;  // dk^-0.5 * log2(e)
    float m2 = -1e30f, l_i = 0.0f;

    stage_chunk(kbb, vtb, 0, &sK[0][0], &sV[0][0], tid);
    wait_stage();
    __syncthreads();

    for (int kt = 0; kt < ATT_T / 32; ++kt) {
        const int cur = kt & 1;
        if (kt + 1 < ATT_T / 32)                // uniform branch
            stage_chunk(kbb, vtb, kt + 1, &sK[cur ^ 1][0], &sV[cur ^ 1][0], tid);

        const unsigned short* kbuf = &sK[cur][0];
        const unsigned short* vbuf = &sV[cur][0];

        const v16bf a00 = load_tile(kbuf + 0 * 16 * SK_STRIDE + 0,  SK_STRIDE);
        const v16bf a01 = load_tile(kbuf + 0 * 16 * SK_STRIDE + 32, SK_STRIDE);
        const v16bf a10 = load_tile(kbuf + 1 * 16 * SK_STRIDE + 0,  SK_STRIDE);
        const v16bf a11 = load_tile(kbuf + 1 * 16 * SK_STRIDE + 32, SK_STRIDE);

        v8f s0 = {}, s1 = {};
        s0 = wmma_bf16(a00, bq0, s0);
        s0 = wmma_bf16(a01, bq1, s0);
        s1 = wmma_bf16(a10, bq0, s1);
        s1 = wmma_bf16(a11, bq1, s1);

        float cmax = -1e30f;
#pragma unroll
        for (int i = 0; i < 8; ++i)
            cmax = fmaxf(cmax, fmaxf(s0[i], s1[i]));
        cmax = fmaxf(cmax, __shfl_xor(cmax, 16, 32));

        const float m_new2 = fmaxf(m2, cmax * cl);
        const float alpha  = EXP2F(m2 - m_new2);

        float p0[8], p1[8];
        float rsum = 0.0f;
#pragma unroll
        for (int i = 0; i < 8; ++i) {
            p0[i] = EXP2F(s0[i] * cl - m_new2);   // fma + v_exp_f32
            p1[i] = EXP2F(s1[i] * cl - m_new2);
            rsum += p0[i] + p1[i];
        }
        rsum += __shfl_xor(rsum, 16, 32);
        l_i = l_i * alpha + rsum;
        m2  = m_new2;

#pragma unroll
        for (int i = 0; i < 8; ++i) { o0[i] *= alpha; o1[i] *= alpha; o2[i] *= alpha; o3[i] *= alpha; }

        ABu bp;
#pragma unroll
        for (int j = 0; j < 4; ++j) {
            bp.ui[j]     = pack_bf16x2(p0[2 * j], p0[2 * j + 1]);
            bp.ui[4 + j] = pack_bf16x2(p1[2 * j], p1[2 * j + 1]);
        }

        const v16bf av0 = load_tile(vbuf + 0 * 16 * SV_STRIDE, SV_STRIDE);
        const v16bf av1 = load_tile(vbuf + 1 * 16 * SV_STRIDE, SV_STRIDE);
        const v16bf av2 = load_tile(vbuf + 2 * 16 * SV_STRIDE, SV_STRIDE);
        const v16bf av3 = load_tile(vbuf + 3 * 16 * SV_STRIDE, SV_STRIDE);
        o0 = wmma_bf16(av0, bp.v, o0);
        o1 = wmma_bf16(av1, bp.v, o1);
        o2 = wmma_bf16(av2, bp.v, o2);
        o3 = wmma_bf16(av3, bp.v, o3);

        wait_stage();                            // next chunk landed in LDS
        __syncthreads();                         // all waves done reading cur
    }

    const float inv = 1.0f / l_i;
    const int n     = lane & 15;
    const int mBase = (lane >> 4) * 8;
    float* obase = out + (size_t)(qtok0 + n) * ATT_DK + mBase;
#pragma unroll
    for (int j = 0; j < 4; ++j) {
        const v8f c = (j == 0) ? o0 : (j == 1) ? o1 : (j == 2) ? o2 : o3;
        float4 lo = {c[0] * inv, c[1] * inv, c[2] * inv, c[3] * inv};
        float4 hi = {c[4] * inv, c[5] * inv, c[6] * inv, c[7] * inv};
        *(float4*)(obase + j * 16)     = lo;
        *(float4*)(obase + j * 16 + 4) = hi;
    }
}

// ---------------------------------------------------------------------------
// Launcher.  Workspace layout (bf16):
//   Wt : 192*1024  bf16  @ 0x000000
//   Q  : 16384*64  bf16  @ 0x060000
//   K  : 16384*64  bf16  @ 0x260000
//   VT : 8*64*2048 bf16  @ 0x460000     total ~6.4 MB
// ---------------------------------------------------------------------------
extern "C" void kernel_launch(void* const* d_in, const int* in_sizes, int n_in,
                              void* d_out, int out_size, void* d_ws, size_t ws_size,
                              hipStream_t stream) {
    const float* x  = (const float*)d_in[0];
    const float* Wq = (const float*)d_in[1];
    const float* Wk = (const float*)d_in[2];
    const float* Wv = (const float*)d_in[3];
    float* out = (float*)d_out;

    char* ws = (char*)d_ws;
    unsigned short* Wt = (unsigned short*)(ws);
    unsigned short* Qb = (unsigned short*)(ws + 0x60000);
    unsigned short* Kb = (unsigned short*)(ws + 0x60000 + 0x200000);
    unsigned short* VT = (unsigned short*)(ws + 0x60000 + 0x400000);

    k_transpose_w<<<dim3(192), dim3(256), 0, stream>>>(Wq, Wk, Wv, Wt);
    k_qkv<<<dim3((ATT_B * ATT_T) / 16), dim3(128), 0, stream>>>(x, Wt, Qb, Kb, VT);
    k_attn<<<dim3((ATT_B * ATT_T) / 16 / 4), dim3(128), 0, stream>>>(Qb, Kb, VT, out);
}
